// LMModel_70136815944215
// MI455X (gfx1250) — matrix operand order repaired and checked by reading, not compiled, not run on hardware
//
#include <hip/hip_runtime.h>
#include <hip/hip_bf16.h>

typedef __attribute__((ext_vector_type(16))) __bf16 v16bf;
typedef __attribute__((ext_vector_type(8)))  float  v8f;

static __device__ __forceinline__ float gelu_tanh(float z) {
    const float c = 0.7978845608028654f;       // sqrt(2/pi)
    float t = tanhf(c * (z + 0.044715f * z * z * z));
    return 0.5f * z * (1.0f + t);
}

// ---------------------------------------------------------------------------
// Embedding gather: h[r,:] = E[x[r,0],:] + E[x[r,1],:]
// ---------------------------------------------------------------------------
__global__ void embed_kernel(const int* __restrict__ x, const float* __restrict__ E,
                             float* __restrict__ h) {
    const int Dc = 768;
    int row = blockIdx.x;
    int i0 = x[row * 2 + 0];
    int i1 = x[row * 2 + 1];
    const float* e0 = E + (size_t)i0 * Dc;
    const float* e1 = E + (size_t)i1 * Dc;
    float* hr = h + (size_t)row * Dc;
#pragma unroll
    for (int i = 0; i < 3; ++i) {
        int c = threadIdx.x + i * 256;
        hr[c] = e0[c] + e1[c];
    }
}

// ---------------------------------------------------------------------------
// bf16 WMMA GEMM, N and K are compile-time so B column loads use one base
// address + immediate offsets. C[MxN] = A[MxK] @ B (+bias, opt GELU, opt B^T).
// Block = 256 threads = 8 wave32; wave w computes a 16x64 tile at
// (blockIdx.y*16, (blockIdx.x*8+w)*64): 4 accumulators share one A fragment.
// Out-of-range rows/cols are CLAMPED (not zero-filled): garbage only reaches
// C rows/cols that are never stored.
// ---------------------------------------------------------------------------
template<int EPI, bool BT, int N, int K>
__global__ void gemm_wmma(const float* __restrict__ A, const float* __restrict__ B,
                          const float* __restrict__ bias, float* __restrict__ C,
                          int M) {
    int wave = threadIdx.x >> 5;
    int lane = threadIdx.x & 31;
    int l15  = lane & 15;
    int hi   = lane >> 4;

    int n0 = (blockIdx.x * 8 + wave) * 64;
    int m0 = blockIdx.y * 16;

    int arow = m0 + l15; if (arow >= M) arow = M - 1;   // clamp, no guard
    const float* arp = A + (size_t)arow * K;

    int bc[4];
#pragma unroll
    for (int s = 0; s < 4; ++s) {
        int c = n0 + s * 16 + l15;
        bc[s] = c < N ? c : N - 1;                      // clamp, no guard
    }

    v8f acc[4];
#pragma unroll
    for (int s = 0; s < 4; ++s) acc[s] = (v8f){0.f,0.f,0.f,0.f,0.f,0.f,0.f,0.f};

    for (int k0 = 0; k0 < K; k0 += 32) {
        // A fragment: lanes 0-15 -> K = k0+[0..15]
        //             lanes 16-31 -> K = k0+16+[0..15]
        const float4* ap = (const float4*)(arp + k0 + hi * 16);
        float4 a0 = ap[0];
        float4 a1 = ap[1];
        const float4* ap2 = (const float4*)(arp + k0 + hi * 16 + 8);
        float4 a2 = ap2[0];
        float4 a3 = ap2[1];
        v16bf va;
        va[0]=(__bf16)a0.x;  va[1]=(__bf16)a0.y;  va[2]=(__bf16)a0.z;  va[3]=(__bf16)a0.w;
        va[4]=(__bf16)a1.x;  va[5]=(__bf16)a1.y;  va[6]=(__bf16)a1.z;  va[7]=(__bf16)a1.w;
        va[8]=(__bf16)a2.x;  va[9]=(__bf16)a2.y;  va[10]=(__bf16)a2.z; va[11]=(__bf16)a2.w;
        va[12]=(__bf16)a3.x; va[13]=(__bf16)a3.y; va[14]=(__bf16)a3.z; va[15]=(__bf16)a3.w;

        int kb = k0 + hi * 16;  // B: lanes 0-15 -> K=k0+[0..15]; lanes 16-31 -> +16
#pragma unroll
        for (int s = 0; s < 4; ++s) {
            float bfv[16];
            if (BT) {
                const float4* bp = (const float4*)(B + (size_t)bc[s] * K + kb);
#pragma unroll
                for (int i = 0; i < 4; ++i) {
                    float4 t = bp[i];
                    bfv[4*i+0] = t.x; bfv[4*i+1] = t.y; bfv[4*i+2] = t.z; bfv[4*i+3] = t.w;
                }
            } else {
                // one base address; i*N becomes an immediate instruction offset
                const float* bp = B + (size_t)kb * N + bc[s];
#pragma unroll
                for (int i = 0; i < 16; ++i) bfv[i] = bp[i * N];
            }
            v16bf vb;
#pragma unroll
            for (int i = 0; i < 16; ++i) vb[i] = (__bf16)bfv[i];
            acc[s] = __builtin_amdgcn_wmma_f32_16x16x32_bf16(
                         false, va, false, vb, (short)0, acc[s], false, false);
        }
    }

    // C layout: VGPR j, lane l -> m = m0 + j + 8*(l>>4), n = tile_n0 + (l&15)
    bool full = (m0 + 16 <= M) && (n0 + 64 <= N);
    if (full) {
#pragma unroll
        for (int s = 0; s < 4; ++s) {
            int n = n0 + s * 16 + l15;
            float bv = bias ? bias[n] : 0.f;
#pragma unroll
            for (int j = 0; j < 8; ++j) {
                int m = m0 + j + 8 * hi;
                float v = acc[s][j] + bv;
                if (EPI == 1) v = gelu_tanh(v);
                C[(size_t)m * N + n] = v;
            }
        }
    } else {
#pragma unroll
        for (int s = 0; s < 4; ++s) {
            int n = n0 + s * 16 + l15;
#pragma unroll
            for (int j = 0; j < 8; ++j) {
                int m = m0 + j + 8 * hi;
                if (m < M && n < N) {
                    float v = acc[s][j];
                    if (bias) v += bias[n];
                    if (EPI == 1) v = gelu_tanh(v);
                    C[(size_t)m * N + n] = v;
                }
            }
        }
    }
}

// ---------------------------------------------------------------------------
// Attention: one wave32 per query row; softmax with causal+pad mask.
// qkv rows: [q(768) | k(768) | v(768)], head slice = head*64.
// ---------------------------------------------------------------------------
__global__ void attn_kernel(const float* __restrict__ qkv, const float* __restrict__ mask,
                            float* __restrict__ out) {
    const int Tc = 625, D3 = 2304, Hc = 12, Dc = 768;
    int wid  = threadIdx.x >> 5;
    int lane = threadIdx.x & 31;
    int q    = blockIdx.x * 4 + wid;
    int bh   = blockIdx.y;
    int b    = bh / Hc;
    int head = bh % Hc;

    bool qval = q < Tc;
    int  qq   = qval ? q : Tc - 1;

    const float* base = qkv + (size_t)b * Tc * D3;
    const float* qrow = base + (size_t)qq * D3 + head * 64;
    float q0 = qrow[lane];
    float q1 = qrow[lane + 32];
    const float* Kp   = base + 768  + head * 64;
    const float* Vp   = base + 1536 + head * 64;
    const float* mrow = mask + (size_t)b * Tc;

    float s[20];
    float mmax = -3.0e38f;
#pragma unroll 4
    for (int j = 0; j < 20; ++j) {
        int k  = j * 32 + lane;
        int kk = k < Tc ? k : Tc - 1;
        const float* kr = Kp + (size_t)kk * D3;
        float acc = 0.f;
        for (int d = 0; d < 32; ++d) acc += __shfl(q0, d) * kr[d];
        for (int d = 0; d < 32; ++d) acc += __shfl(q1, d) * kr[d + 32];
        bool valid = (k <= qq) && (k < Tc) && (mrow[kk] != 0.f);
        s[j] = valid ? acc * 0.125f : -1.0e30f;
        mmax = fmaxf(mmax, s[j]);
    }
#pragma unroll
    for (int off = 16; off > 0; off >>= 1) mmax = fmaxf(mmax, __shfl_xor(mmax, off));

    float ssum = 0.f;
#pragma unroll
    for (int j = 0; j < 20; ++j) {
        s[j] = expf(s[j] - mmax);
        ssum += s[j];
    }
#pragma unroll
    for (int off = 16; off > 0; off >>= 1) ssum += __shfl_xor(ssum, off);
    float inv = 1.0f / ssum;
#pragma unroll
    for (int j = 0; j < 20; ++j) s[j] *= inv;

    float a0 = 0.f, a1 = 0.f;
    for (int j = 0; j < 20; ++j) {
        int kbase = j * 32;
        if (kbase > qq) break;                 // uniform across wave
        for (int t = 0; t < 32; ++t) {
            int k = kbase + t;
            if (k > qq) break;                 // uniform across wave
            float w = __shfl(s[j], t);
            const float* vr = Vp + (size_t)k * D3;
            a0 += w * vr[lane];
            a1 += w * vr[lane + 32];
        }
    }
    if (qval) {
        float* orow = out + ((size_t)b * Tc + q) * Dc + head * 64;
        orow[lane]      = a0;
        orow[lane + 32] = a1;
    }
}

// ---------------------------------------------------------------------------
// out[row,:] = LayerNorm(a[row,:] + b[row,:]; g, beta), D = 768, eps = 1e-5
// ---------------------------------------------------------------------------
__global__ void add_ln_kernel(float* __restrict__ out, const float* __restrict__ a,
                              const float* __restrict__ bb, const float* __restrict__ g,
                              const float* __restrict__ beta) {
    const int Dc = 768;
    int row  = blockIdx.x;
    int tid  = threadIdx.x;
    int wid  = tid >> 5;
    int lane = tid & 31;
    __shared__ float rs[8], rq[8];

    float v[3];
    float sum = 0.f, sq = 0.f;
#pragma unroll
    for (int i = 0; i < 3; ++i) {
        int c = tid + i * 256;
        float z = a[(size_t)row * Dc + c] + bb[(size_t)row * Dc + c];
        v[i] = z;
        sum += z;
        sq  += z * z;
    }
#pragma unroll
    for (int off = 16; off > 0; off >>= 1) {
        sum += __shfl_xor(sum, off);
        sq  += __shfl_xor(sq,  off);
    }
    if (lane == 0) { rs[wid] = sum; rq[wid] = sq; }
    __syncthreads();
    if (tid == 0) {
        float S = 0.f, Q = 0.f;
#pragma unroll
        for (int i = 0; i < 8; ++i) { S += rs[i]; Q += rq[i]; }
        rs[0] = S; rq[0] = Q;
    }
    __syncthreads();
    float mean = rs[0] * (1.0f / Dc);
    float var  = rq[0] * (1.0f / Dc) - mean * mean;
    float rstd = rsqrtf(var + 1e-5f);
#pragma unroll
    for (int i = 0; i < 3; ++i) {
        int c = tid + i * 256;
        out[(size_t)row * Dc + c] = g[c] * (v[i] - mean) * rstd + beta[c];
    }
}

// ---------------------------------------------------------------------------
extern "C" void kernel_launch(void* const* d_in, const int* in_sizes, int n_in,
                              void* d_out, int out_size, void* d_ws, size_t ws_size,
                              hipStream_t stream) {
    (void)in_sizes; (void)n_in; (void)out_size; (void)ws_size;
    const int Lc = 12, Hc = 12, Dc = 768, DFFc = 3072, Vc = 40990, Tc = 625, Bc = 2;
    const int M = Bc * Tc;  // 1250

    const int*   x      = (const int*)  d_in[0];
    const float* mask   = (const float*)d_in[1];
    const float* E      = (const float*)d_in[2];
    const float* attn_w = (const float*)d_in[3];
    const float* attn_b = (const float*)d_in[4];
    const float* proj_w = (const float*)d_in[5];
    const float* proj_b = (const float*)d_in[6];
    const float* ln1_g  = (const float*)d_in[7];
    const float* ln1_b  = (const float*)d_in[8];
    const float* fc_w   = (const float*)d_in[9];
    const float* fc_b   = (const float*)d_in[10];
    const float* fc2_w  = (const float*)d_in[11];
    const float* fc2_b  = (const float*)d_in[12];
    const float* ln2_g  = (const float*)d_in[13];
    const float* ln2_b  = (const float*)d_in[14];
    float* out = (float*)d_out;

    float* w     = (float*)d_ws;
    float* hbuf  = w; w += (size_t)M * Dc;
    float* qkv   = w; w += (size_t)M * 3 * Dc;
    float* attnb = w; w += (size_t)M * Dc;
    float* tmpb  = w; w += (size_t)M * Dc;
    float* nbuf  = w; w += (size_t)M * Dc;
    float* fbuf  = w; w += (size_t)M * DFFc;      // total ~42 MB of d_ws

    dim3 blk(256);
    dim3 g_qkv((3 * Dc + 511) / 512, (M + 15) / 16);
    dim3 g_d  ((Dc + 511) / 512,     (M + 15) / 16);
    dim3 g_ff ((DFFc + 511) / 512,   (M + 15) / 16);
    dim3 g_lm ((Vc + 511) / 512,     (M + 15) / 16);
    dim3 g_at ((Tc + 3) / 4, Bc * Hc);

    embed_kernel<<<M, blk, 0, stream>>>(x, E, hbuf);

    for (int l = 0; l < Lc; ++l) {
        gemm_wmma<0, false, 2304, 768><<<g_qkv, blk, 0, stream>>>(
            hbuf, attn_w + (size_t)l * Dc * 3 * Dc, attn_b + (size_t)l * 3 * Dc,
            qkv, M);
        attn_kernel<<<g_at, dim3(128), 0, stream>>>(qkv, mask, attnb);
        gemm_wmma<0, false, 768, 768><<<g_d, blk, 0, stream>>>(
            attnb, proj_w + (size_t)l * Dc * Dc, proj_b + (size_t)l * Dc,
            tmpb, M);
        add_ln_kernel<<<M, blk, 0, stream>>>(nbuf, hbuf, tmpb,
                                             ln1_g + (size_t)l * Dc, ln1_b + (size_t)l * Dc);
        gemm_wmma<1, false, 3072, 768><<<g_ff, blk, 0, stream>>>(
            nbuf, fc_w + (size_t)l * Dc * DFFc, fc_b + (size_t)l * DFFc,
            fbuf, M);
        gemm_wmma<0, false, 768, 3072><<<g_d, blk, 0, stream>>>(
            fbuf, fc2_w + (size_t)l * DFFc * Dc, fc2_b + (size_t)l * Dc,
            tmpb, M);
        add_ln_kernel<<<M, blk, 0, stream>>>(hbuf, nbuf, tmpb,
                                             ln2_g + (size_t)l * Dc, ln2_b + (size_t)l * Dc);
    }

    // logits = h @ E^T   (E is [V, D] row-major -> B^T mode: contiguous K per column)
    gemm_wmma<0, true, 40990, 768><<<g_lm, blk, 0, stream>>>(hbuf, E, nullptr, out, M);
}
